// InferenceNet_10118942949387
// MI455X (gfx1250) — matrix operand order repaired and checked by reading, not compiled, not run on hardware
//
#include <hip/hip_runtime.h>
#include <hip/hip_bf16.h>

// ---------------------------------------------------------------------------
// Problem dims (from reference): B=8, T=8192, IDIM=ODIM=256, HDIM=1024, CDIM=128
// Inputs (d_in order): x, mask_prev, W_enc, b_enc, W_dec_self, b_dec_self,
//                      W_dec_src, b_dec_src, decoder_type
// Outputs (d_out): out [B*T*256] f32, then mask_new [B*T*1024] f32
// ---------------------------------------------------------------------------

typedef __attribute__((ext_vector_type(16))) __bf16 v16bf;
typedef __attribute__((ext_vector_type(8)))  __bf16 v8bf;
typedef __attribute__((ext_vector_type(8)))  float  v8f;

constexpr int kB    = 8;
constexpr int kT    = 8192;
constexpr int IDIM  = 256;
constexpr int ODIM  = 256;
constexpr int HDIM  = 1024;
constexpr int CDIM  = 128;
constexpr int NTOK  = kB * kT;      // 65536 tokens
constexpr int ROWS  = 16;           // tokens per workgroup (one WMMA M-tile)
constexpr int HPAD  = HDIM + 8;     // LDS row padding

// magnitude key of bf16 value: monotonic in h^2 (sign cleared)
__device__ __forceinline__ unsigned keyOf(__bf16 h) {
    return (unsigned)(__builtin_bit_cast(unsigned short, h) & 0x7FFFu);
}

// ------------------------- weight prep kernels -----------------------------

// W_enc [K=256][N=1024] f32  ->  WencT [N=1024][K=256] bf16
__global__ void prep_wenc(const float* __restrict__ W, __bf16* __restrict__ out) {
    int i = blockIdx.x * 256 + threadIdx.x;      // over IDIM*HDIM
    int k = i >> 10, n = i & (HDIM - 1);
    out[(long)n * IDIM + k] = (__bf16)W[i];
}

// W_dec [K=1024][N=256] f32 (self or src picked by decoder_type) -> WdecT [N][K] bf16
__global__ void prep_wdec(const float* __restrict__ Wself, const float* __restrict__ Wsrc,
                          const int* __restrict__ dtype, __bf16* __restrict__ out) {
    const float* W = (*dtype == 0) ? Wself : Wsrc;
    int i = blockIdx.x * 256 + threadIdx.x;      // over HDIM*ODIM
    int k = i >> 8, n = i & (ODIM - 1);
    out[(long)n * HDIM + k] = (__bf16)W[i];
}

__global__ void prep_bdec(const float* __restrict__ bself, const float* __restrict__ bsrc,
                          const int* __restrict__ dtype, float* __restrict__ out) {
    int i = threadIdx.x;
    out[i] = (*dtype == 0) ? bself[i] : bsrc[i];
}

// ------------------------------ fused kernel -------------------------------

__global__ __launch_bounds__(256)
void fused_enc_topk_dec(const float* __restrict__ x,
                        const float* __restrict__ mask_prev,
                        const float* __restrict__ b_enc,
                        const __bf16* __restrict__ WencT,
                        const __bf16* __restrict__ WdecT,
                        const float* __restrict__ b_dec,
                        float* __restrict__ out,
                        float* __restrict__ mask_new) {
    __shared__ __align__(32) __bf16 sXA[ROWS][IDIM];     // 8 KB  (A operand, encoder)
    __shared__ __align__(32) __bf16 sH[ROWS][HPAD];      // 33 KB (h tile, then masked h)
    __shared__ unsigned sHist[8][256];                   // 8 KB  (per-wave radix hist)
    __shared__ unsigned sScan[8][4];
    __shared__ unsigned sT_cur[ROWS], sBud_cur[ROWS], sT_share[ROWS], sBud_share[ROWS];
    __shared__ unsigned sTieC[ROWS], sTieS[ROWS];

    const int  tid  = threadIdx.x;
    const int  wave = tid >> 5;
    const int  lane = tid & 31;
    const long tok0 = (long)blockIdx.x * ROWS;

    // ---- load x tile, convert f32 -> bf16 into A layout (row-major, K contig)
    #pragma unroll
    for (int i = 0; i < ROWS * IDIM / 256; ++i) {
        int idx = tid + i * 256;
        sXA[idx >> 8][idx & (IDIM - 1)] = (__bf16)x[tok0 * IDIM + idx];
    }
    __syncthreads();

    // ---- encoder: h[16][1024] = x[16][256] * WencT^T, via WMMA bf16
    {
        const int m  = lane & 15;     // A row (token)
        const int kh = lane >> 4;     // K half selector
        #pragma unroll
        for (int nt = 0; nt < (HDIM / 16) / 8; ++nt) {   // 8 n-tiles per wave
            const int n0   = (wave * 8 + nt) * 16;
            const int ncol = n0 + (lane & 15);
            const __bf16* wcol = WencT + (long)ncol * IDIM;
            v8f acc = {};
            #pragma unroll
            for (int ks = 0; ks < IDIM / 32; ++ks) {
                const int k0 = ks * 32;
                v8bf a0 = *(const v8bf*)&sXA[m][k0 + 8 * kh];
                v8bf a1 = *(const v8bf*)&sXA[m][k0 + 16 + 8 * kh];
                v16bf a = __builtin_shufflevector(a0, a1, 0,1,2,3,4,5,6,7,8,9,10,11,12,13,14,15);
                v8bf b0 = *(const v8bf*)(wcol + k0 + 8 * kh);
                v8bf b1 = *(const v8bf*)(wcol + k0 + 16 + 8 * kh);
                v16bf b = __builtin_shufflevector(b0, b1, 0,1,2,3,4,5,6,7,8,9,10,11,12,13,14,15);
                acc = __builtin_amdgcn_wmma_f32_16x16x32_bf16(
                          false, a, false, b, (short)0, acc, false, false);
            }
            const float bias = b_enc[ncol];
            #pragma unroll
            for (int r = 0; r < 8; ++r) {               // C layout: M = r + 8*kh
                const int row = r + 8 * kh;
                const float mp = mask_prev[(tok0 + row) * HDIM + ncol];
                const float hv = (mp > 0.0f) ? 0.0f : (acc[r] + bias);
                sH[row][ncol] = (__bf16)hv;
            }
        }
    }
    __syncthreads();

    // ---- exact top-k threshold per token via 2-level radix select on |h| key
    // one wave per token; 2 tokens per wave; all barriers workgroup-uniform
    for (int it = 0; it < 2; ++it) {
        const int t = wave * 2 + it;
        unsigned* hist = sHist[wave];
        for (int i2 = lane; i2 < 256; i2 += 32) hist[i2] = 0;
        __syncthreads();
        #pragma unroll 4
        for (int i2 = 0; i2 < HDIM / 32; ++i2) {
            unsigned key = keyOf(sH[t][lane + i2 * 32]);
            atomicAdd(&hist[key >> 7], 1u);             // top 8 bits of 15-bit key
        }
        __syncthreads();
        if (lane == 0) {
            unsigned cum = 0, rc = 0, rs = 0;
            int bc = -1, bs = -1;
            for (int b = 255; b >= 0; --b) {
                unsigned c = hist[b];
                if (bc < 0 && cum + c >= (unsigned)CDIM)     { bc = b; rc = CDIM - cum; }
                if (bs < 0 && cum + c >= (unsigned)(2*CDIM)) { bs = b; rs = 2*CDIM - cum; break; }
                cum += c;
            }
            sScan[wave][0] = (unsigned)bc; sScan[wave][1] = rc;
            sScan[wave][2] = (unsigned)bs; sScan[wave][3] = rs;
            sTieC[t] = 0; sTieS[t] = 0;
        }
        __syncthreads();
        #pragma unroll
        for (int which = 0; which < 2; ++which) {       // 0: k=128 (cur), 1: k=256 (share)
            const unsigned b1  = sScan[wave][which * 2 + 0];
            const unsigned rem = sScan[wave][which * 2 + 1];
            for (int i2 = lane; i2 < 128; i2 += 32) hist[i2] = 0;
            __syncthreads();
            #pragma unroll 4
            for (int i2 = 0; i2 < HDIM / 32; ++i2) {
                unsigned key = keyOf(sH[t][lane + i2 * 32]);
                if ((key >> 7) == b1) atomicAdd(&hist[key & 127], 1u);  // low 7 bits
            }
            __syncthreads();
            if (lane == 0) {
                unsigned cum = 0, c2 = 0, bud = rem;
                for (int b = 127; b >= 0; --b) {
                    unsigned c = hist[b];
                    if (cum + c >= rem) { c2 = (unsigned)b; bud = rem - cum; break; }
                    cum += c;
                }
                const unsigned T = (b1 << 7) | c2;
                if (which == 0) { sT_cur[t]   = T; sBud_cur[t]   = bud; }
                else            { sT_share[t] = T; sBud_share[t] = bud; }
            }
            __syncthreads();
        }
    }

    // ---- apply masks: write mask_new (coalesced), zero non-top-256 h in LDS
    #pragma unroll 4
    for (int rr = 0; rr < ROWS * HDIM / 256; ++rr) {
        const int flat = rr * 256 + tid;
        const int t = flat >> 10, j = flat & (HDIM - 1);
        const unsigned key = keyOf(sH[t][j]);
        const unsigned Ts  = sT_share[t];
        bool selS = key > Ts || (key == Ts && atomicAdd(&sTieS[t], 1u) < sBud_share[t]);
        bool selC = false;
        if (selS) {
            const unsigned Tc = sT_cur[t];
            selC = key > Tc || (key == Tc && atomicAdd(&sTieC[t], 1u) < sBud_cur[t]);
        }
        const long gm = (tok0 + t) * HDIM + j;
        mask_new[gm] = mask_prev[gm] + (selC ? 1.0f : 0.0f);
        if (!selS) sH[t][j] = (__bf16)0.0f;
    }
    __syncthreads();

    // ---- decoder: out[16][256] = h_masked[16][1024] * WdecT^T, via WMMA bf16
    {
        const int m  = lane & 15;
        const int kh = lane >> 4;
        #pragma unroll
        for (int nt = 0; nt < (ODIM / 16) / 8; ++nt) {  // 2 n-tiles per wave
            const int n0   = (wave * 2 + nt) * 16;
            const int ncol = n0 + (lane & 15);
            const __bf16* wcol = WdecT + (long)ncol * HDIM;
            v8f acc = {};
            for (int ks = 0; ks < HDIM / 32; ++ks) {
                const int k0 = ks * 32;
                __builtin_prefetch(wcol + k0 + 128, 0, 1);   // global_prefetch_b8
                v8bf a0 = *(const v8bf*)&sH[m][k0 + 8 * kh];
                v8bf a1 = *(const v8bf*)&sH[m][k0 + 16 + 8 * kh];
                v16bf a = __builtin_shufflevector(a0, a1, 0,1,2,3,4,5,6,7,8,9,10,11,12,13,14,15);
                v8bf b0 = *(const v8bf*)(wcol + k0 + 8 * kh);
                v8bf b1 = *(const v8bf*)(wcol + k0 + 16 + 8 * kh);
                v16bf b = __builtin_shufflevector(b0, b1, 0,1,2,3,4,5,6,7,8,9,10,11,12,13,14,15);
                acc = __builtin_amdgcn_wmma_f32_16x16x32_bf16(
                          false, a, false, b, (short)0, acc, false, false);
            }
            const float bias = b_dec[ncol];
            #pragma unroll
            for (int r = 0; r < 8; ++r) {
                const int row = r + 8 * kh;
                out[(tok0 + row) * ODIM + ncol] = acc[r] + bias;
            }
        }
    }
}

// ------------------------------ host launcher ------------------------------

extern "C" void kernel_launch(void* const* d_in, const int* in_sizes, int n_in,
                              void* d_out, int out_size, void* d_ws, size_t ws_size,
                              hipStream_t stream) {
    const float* x          = (const float*)d_in[0];
    const float* mask_prev  = (const float*)d_in[1];
    const float* W_enc      = (const float*)d_in[2];
    const float* b_enc      = (const float*)d_in[3];
    const float* W_dec_self = (const float*)d_in[4];
    const float* b_dec_self = (const float*)d_in[5];
    const float* W_dec_src  = (const float*)d_in[6];
    const float* b_dec_src  = (const float*)d_in[7];
    const int*   dtype      = (const int*)d_in[8];

    float* out      = (float*)d_out;
    float* mask_new = out + (size_t)NTOK * ODIM;

    char* ws = (char*)d_ws;
    __bf16* WencT = (__bf16*)ws;                                   // 512 KB
    __bf16* WdecT = (__bf16*)(ws + (size_t)IDIM * HDIM * 2);       // 512 KB
    float*  bdec  = (float*)(ws + (size_t)IDIM * HDIM * 2 + (size_t)HDIM * ODIM * 2);

    prep_wenc<<<IDIM * HDIM / 256, 256, 0, stream>>>(W_enc, WencT);
    prep_wdec<<<HDIM * ODIM / 256, 256, 0, stream>>>(W_dec_self, W_dec_src, dtype, WdecT);
    prep_bdec<<<1, ODIM, 0, stream>>>(b_dec_self, b_dec_src, dtype, bdec);

    fused_enc_topk_dec<<<NTOK / ROWS, 256, 0, stream>>>(
        x, mask_prev, b_enc, WencT, WdecT, bdec, out, mask_new);
}
